// ALIGNNAtomWise_69466801045956
// MI455X (gfx1250) — compile-verified, matrix operand order checked by reading, not run on hardware
//
#include <hip/hip_runtime.h>
#include <cstddef>
#include <cstdint>

#define HID 256

typedef __attribute__((ext_vector_type(16))) __bf16 v16bf;
typedef __attribute__((ext_vector_type(8)))  float  v8f;

union FragAB { v16bf v; unsigned u[8]; };

__device__ __forceinline__ unsigned short f2bf(float f) {
    unsigned u = __float_as_uint(f);
    u += 0x7FFFu + ((u >> 16) & 1u);           // round-to-nearest-even
    return (unsigned short)(u >> 16);
}

__device__ __forceinline__ unsigned lds_addr_of(const void* p) {
    // generic pointer to LDS: low 32 bits hold the LDS byte offset
    return (unsigned)(unsigned long long)p;
}

__device__ __forceinline__ void async_copy_b64(unsigned lds, const void* gaddr) {
    asm volatile("global_load_async_to_lds_b64 %0, %1, off"
                 :: "v"(lds), "v"(gaddr) : "memory");
}
__device__ __forceinline__ void async_copy_b128(unsigned lds, const void* gaddr) {
    asm volatile("global_load_async_to_lds_b128 %0, %1, off"
                 :: "v"(lds), "v"(gaddr) : "memory");
}
__device__ __forceinline__ void wait_async0() {
    asm volatile("s_wait_asynccnt 0" ::: "memory");
}

// ---------------------------------------------------------------------------
// Tiled WMMA GEMM: out[M,N] = A[M,Kp](bf16) x Wt[N,Kp](bf16, pre-transposed)
// + bias, optional fused LayerNorm+SiLU epilogue (MODE==1 -> f32 + bf16).
// Block = 256 threads (8 waves), BM=32 rows, BN=N columns (full width).
// Tiles staged with global_load_async_to_lds (ASYNCcnt), fragments read as
// ds_load_b128 pairs, math on v_wmma_f32_16x16x32_bf16.
// ---------------------------------------------------------------------------
template <int N, int MODE>
__global__ __launch_bounds__(256) void k_gemm(
    const unsigned short* __restrict__ A, const unsigned short* __restrict__ Wt,
    const float* __restrict__ bias, const float* __restrict__ gamma,
    const float* __restrict__ beta, float* __restrict__ outF,
    unsigned short* __restrict__ outB, int M, int Kp) {
    constexpr int BM    = 32;
    constexpr int NUMTN = N / 16;
    constexpr int TPW   = (2 * NUMTN) / 8;       // tiles per wave (1 or 4)

    __shared__ unsigned short As[BM][32];        // A tile, row-major
    __shared__ unsigned short Bst[N][32];        // B tile, [n][k] (as in Wt)
    __shared__ float Cs[BM][N];
    __shared__ float s_mu[BM], s_ri[BM];

    const int tid  = threadIdx.x;
    const int wave = tid >> 5;
    const int lane = tid & 31;
    const int m0   = blockIdx.x * BM;
    const int tmBase = ((wave * TPW) / NUMTN) * 16;

    // per-thread staging coordinates (whole tile covered by 256 threads)
    const int arow = tid >> 3;                   // A: 8 bytes (4 bf16) each
    const int acol = (tid & 7) * 4;
    const unsigned aLds = lds_addr_of(&As[arow][acol]);

    v8f acc[TPW] = {};

    for (int k0 = 0; k0 < Kp; k0 += 32) {
        // ---- async stage A tile: 32x32 bf16, contiguous 8B chunks ----
        async_copy_b64(aLds, &A[(size_t)(m0 + arow) * Kp + k0 + acol]);
        if (k0 + 32 < Kp)                        // -> global_prefetch_b8
            __builtin_prefetch(&A[(size_t)(m0 + arow) * Kp + k0 + 32 + acol], 0, 0);
        // ---- async stage B tile: N x 32 bf16 from pre-transposed Wt ----
#pragma unroll
        for (int i = 0; i < N / 64; i++) {
            const int id   = tid + i * 256;      // 16B chunk id, 4N total
            const int n    = id >> 2;
            const int koff = (id & 3) * 8;
            async_copy_b128(lds_addr_of(&Bst[n][koff]),
                            &Wt[(size_t)n * Kp + k0 + koff]);
        }
        wait_async0();
        __syncthreads();

        // ---- A fragment (16x32 bf16, lane = M row) ----
        FragAB fa;
        {
            const int m    = tmBase + (lane & 15);
            const int half = lane >> 4;
#pragma unroll
            for (int i = 0; i < 4; i++)
                fa.u[i] = *reinterpret_cast<const unsigned*>(&As[m][half * 8 + 2 * i]);
#pragma unroll
            for (int i = 0; i < 4; i++)
                fa.u[4 + i] = *reinterpret_cast<const unsigned*>(&As[m][16 + half * 8 + 2 * i]);
        }
#pragma unroll
        for (int t = 0; t < TPW; t++) {
            const int tn   = (wave * TPW + t) % NUMTN;
            const int col  = tn * 16 + (lane & 15);
            const int half = lane >> 4;
            FragAB fb;
#pragma unroll
            for (int i = 0; i < 4; i++)
                fb.u[i] = *reinterpret_cast<const unsigned*>(&Bst[col][half * 8 + 2 * i]);
#pragma unroll
            for (int i = 0; i < 4; i++)
                fb.u[4 + i] = *reinterpret_cast<const unsigned*>(&Bst[col][16 + half * 8 + 2 * i]);
            acc[t] = __builtin_amdgcn_wmma_f32_16x16x32_bf16(
                false, fa.v, false, fb.v, (short)0, acc[t], false, false);
        }
        __syncthreads();
    }

    // ---- spill accumulators to LDS per the 16x16 f32 C/D layout ----
#pragma unroll
    for (int t = 0; t < TPW; t++) {
        const int tn  = (wave * TPW + t) % NUMTN;
        const int col = tn * 16 + (lane & 15);
#pragma unroll
        for (int r = 0; r < 8; r++) {
            const int mrow = tmBase + ((lane >> 4) ? (r + 8) : r);
            Cs[mrow][col] = acc[t][r];
        }
    }
    __syncthreads();

    if constexpr (MODE == 1) {
        if (tid < BM) {
            float s = 0.f, q = 0.f;
            for (int n = 0; n < N; n++) {
                const float v = Cs[tid][n] + bias[n];
                s += v; q += v * v;
            }
            const float mu = s / N;
            s_mu[tid] = mu;
            s_ri[tid] = rsqrtf(q / N - mu * mu + 1e-5f);
        }
        __syncthreads();
#pragma unroll
        for (int i = 0; i < (BM * N) / 256; i++) {
            const int e = tid + i * 256;
            const int row = e / N, col = e % N;
            const float v = Cs[row][col] + bias[col];
            const float a = (v - s_mu[row]) * s_ri[row] * gamma[col] + beta[col];
            const float o = a * (1.0f / (1.0f + __expf(-a)));   // SiLU
            const size_t gi = (size_t)(m0 + row) * N + col;
            if (outF) outF[gi] = o;
            outB[gi] = f2bf(o);
        }
    } else {
#pragma unroll
        for (int i = 0; i < (BM * N) / 256; i++) {
            const int e = tid + i * 256;
            const int row = e / N, col = e % N;
            outF[(size_t)(m0 + row) * N + col] = Cs[row][col] + bias[col];
        }
    }
}

// ---------------------------------------------------------------------------
// EGGC edge step: m = Gs[src]+Gd[dst]+Ce ; sigma=sigmoid(m);
// scatter-add Bh[src]*sigma and sigma into dst; y += silu(LN(m)).
// One wave32 per edge row of HID=256 (8 cols per lane).
// ---------------------------------------------------------------------------
__global__ __launch_bounds__(256) void k_edge(
    const float* __restrict__ Gs, const float* __restrict__ Gd,
    const float* __restrict__ Bh, const float* __restrict__ Ce,
    const int* __restrict__ src, const int* __restrict__ dst,
    const float* __restrict__ lng, const float* __restrict__ lnb,
    float* __restrict__ y, unsigned short* __restrict__ yb,
    float* __restrict__ sum_h, float* __restrict__ sum_s, int nE) {
    const int e = blockIdx.x * 8 + (threadIdx.x >> 5);
    if (e >= nE) return;
    const int lane = threadIdx.x & 31;
    const size_t eb = (size_t)e * HID;
    const size_t sb = (size_t)src[e] * HID;
    const size_t db = (size_t)dst[e] * HID;

    float m[8];
    float sum = 0.f, sq = 0.f;
#pragma unroll
    for (int j = 0; j < 8; j++) {
        const int c = lane + 32 * j;
        const float v = Gs[sb + c] + Gd[db + c] + Ce[eb + c];
        m[j] = v; sum += v; sq += v * v;
    }
#pragma unroll
    for (int o = 16; o > 0; o >>= 1) {
        sum += __shfl_xor(sum, o, 32);
        sq  += __shfl_xor(sq, o, 32);
    }
    const float mu = sum / HID;
    const float ri = rsqrtf(sq / HID - mu * mu + 1e-5f);
#pragma unroll
    for (int j = 0; j < 8; j++) {
        const int c = lane + 32 * j;
        const float sig = 1.0f / (1.0f + __expf(-m[j]));
        atomicAdd(&sum_h[db + c], Bh[sb + c] * sig);
        atomicAdd(&sum_s[db + c], sig);
        const float a  = (m[j] - mu) * ri * lng[c] + lnb[c];
        const float yn = y[eb + c] + a * (1.0f / (1.0f + __expf(-a)));
        y[eb + c]  = yn;
        yb[eb + c] = f2bf(yn);
    }
}

// EGGC node step: h = sum_h/(sum_s+1e-6); x += silu(LN(U + h)).
__global__ __launch_bounds__(256) void k_node(
    const float* __restrict__ U, const float* __restrict__ sum_h,
    const float* __restrict__ sum_s, const float* __restrict__ lng,
    const float* __restrict__ lnb, float* __restrict__ x,
    unsigned short* __restrict__ xb, int nN) {
    const int r = blockIdx.x * 8 + (threadIdx.x >> 5);
    if (r >= nN) return;
    const int lane = threadIdx.x & 31;
    const size_t rb = (size_t)r * HID;
    float t[8];
    float sum = 0.f, sq = 0.f;
#pragma unroll
    for (int j = 0; j < 8; j++) {
        const int c = lane + 32 * j;
        const float h = sum_h[rb + c] / (sum_s[rb + c] + 1e-6f);
        const float v = U[rb + c] + h;
        t[j] = v; sum += v; sq += v * v;
    }
#pragma unroll
    for (int o = 16; o > 0; o >>= 1) {
        sum += __shfl_xor(sum, o, 32);
        sq  += __shfl_xor(sq, o, 32);
    }
    const float mu = sum / HID;
    const float ri = rsqrtf(sq / HID - mu * mu + 1e-5f);
#pragma unroll
    for (int j = 0; j < 8; j++) {
        const int c = lane + 32 * j;
        const float a  = (t[j] - mu) * ri * lng[c] + lnb[c];
        const float xn = x[rb + c] + a * (1.0f / (1.0f + __expf(-a)));
        x[rb + c]  = xn;
        xb[rb + c] = f2bf(xn);
    }
}

// RBF expansions -> bf16 padded rows
__global__ void k_rbf_edge(const float* __restrict__ r,
                           unsigned short* __restrict__ out, int nE) {
    const int e = blockIdx.x * 256 + threadIdx.x;
    if (e >= nE) return;
    const float a = r[e * 3], b = r[e * 3 + 1], c = r[e * 3 + 2];
    const float d = sqrtf(a * a + b * b + c * c);
    const float gamma = 79.0f / 8.0f, step = 8.0f / 79.0f;
    for (int i = 0; i < 80; i++) {
        const float dd = d - step * i;
        out[(size_t)e * 96 + i] = f2bf(__expf(-gamma * dd * dd));
    }
    for (int i = 80; i < 96; i++) out[(size_t)e * 96 + i] = 0;
}

__global__ void k_rbf_angle(const float* __restrict__ ah,
                            unsigned short* __restrict__ out, int nT) {
    const int t = blockIdx.x * 256 + threadIdx.x;
    if (t >= nT) return;
    const float d = ah[t];
    const float gamma = 39.0f / 2.0f, step = 2.0f / 39.0f;
    for (int i = 0; i < 40; i++) {
        const float dd = d - (-1.0f + step * i);
        out[(size_t)t * 64 + i] = f2bf(__expf(-gamma * dd * dd));
    }
    for (int i = 40; i < 64; i++) out[(size_t)t * 64 + i] = 0;
}

// f32 weight [K,N] -> bf16 TRANSPOSED padded Wt[N,Kp] (zero pad k >= K)
__global__ void k_cvtW(const float* __restrict__ W, unsigned short* __restrict__ Wt,
                       int K, int Kp, int Ncols) {
    const int id = blockIdx.x * 256 + threadIdx.x;
    if (id >= Kp * Ncols) return;
    const int n = id / Kp, k = id % Kp;
    Wt[id] = (k < K) ? f2bf(W[(size_t)k * Ncols + n]) : (unsigned short)0;
}

// f32 activation [M,K] -> bf16 padded [M,Kp] (zero pad cols)
__global__ void k_cvtA(const float* __restrict__ A, unsigned short* __restrict__ Ap,
                       int M, int K, int Kp) {
    const int id = blockIdx.x * 256 + threadIdx.x;
    if (id >= M * Kp) return;
    const int m = id / Kp, k = id % Kp;
    Ap[id] = (k < K) ? f2bf(A[(size_t)m * K + k]) : (unsigned short)0;
}

__global__ void k_zero(float* __restrict__ p, size_t n) {
    const size_t i = (size_t)blockIdx.x * 256 + threadIdx.x;
    if (i < n) p[i] = 0.0f;
}

// mean over nodes then dot with fc.W + fc.b -> scalar
__global__ __launch_bounds__(256) void k_meanfc(const float* __restrict__ x,
                                                const float* __restrict__ W,
                                                const float* __restrict__ b,
                                                float* __restrict__ out, int nN) {
    __shared__ float red[256];
    const int tid = threadIdx.x;
    float s = 0.f;
    for (int r = 0; r < nN; r++) s += x[(size_t)r * HID + tid];
    red[tid] = (s / nN) * W[tid];
    __syncthreads();
    for (int o = 128; o > 0; o >>= 1) {
        if (tid < o) red[tid] += red[tid + o];
        __syncthreads();
    }
    if (tid == 0) out[0] = red[0] + b[0];
}

// ---------------------------------------------------------------------------
extern "C" void kernel_launch(void* const* d_in, const int* in_sizes, int n_in,
                              void* d_out, int out_size, void* d_ws, size_t ws_size,
                              hipStream_t stream) {
    const int NN = 4096, NE = 65536, NT = 262144;
    if (n_in < 197) return;  // unexpected param flattening; avoid OOB host reads

    auto F = [&](int i) { return (const float*)d_in[i]; };
    auto I = [&](int i) { return (const int*)d_in[i]; };

    size_t off = 0;
    auto alloc = [&](size_t bytes) -> void* {
        off = (off + 255) & ~(size_t)255;
        void* p = (char*)d_ws + off;
        off += bytes;
        return p;
    };

    // activations: f32 + bf16 shadow
    unsigned short* xb = (unsigned short*)alloc((size_t)NN * HID * 2);
    float*          xf = (float*)alloc((size_t)NN * HID * 4);
    unsigned short* yb = (unsigned short*)alloc((size_t)NE * HID * 2);
    float*          yf = (float*)alloc((size_t)NE * HID * 4);
    float*          zf = (float*)alloc((size_t)NT * HID * 4);
    unsigned short* zb = (unsigned short*)alloc((size_t)NT * HID * 2);
    // gate buffers (sized for line-graph rowsX = NE) + edge gate (rowsY = NT)
    float* g0 = (float*)alloc((size_t)NE * HID * 4);
    float* g1 = (float*)alloc((size_t)NE * HID * 4);
    float* g2 = (float*)alloc((size_t)NE * HID * 4);
    float* g3 = (float*)alloc((size_t)NE * HID * 4);
    float* ce = (float*)alloc((size_t)NT * HID * 4);
    float* sh = (float*)alloc((size_t)NE * HID * 4);
    float* ss = (float*)alloc((size_t)NE * HID * 4);
    // feature buffers
    unsigned short* atomb = (unsigned short*)alloc((size_t)NN * 96 * 2);
    unsigned short* rbfe  = (unsigned short*)alloc((size_t)NE * 96 * 2);
    unsigned short* e1b   = (unsigned short*)alloc((size_t)NE * 64 * 2);
    unsigned short* rbfa  = (unsigned short*)alloc((size_t)NT * 64 * 2);
    unsigned short* a1b   = (unsigned short*)alloc((size_t)NT * 64 * 2);

    auto cvtW = [&](int leaf, int K, int Kp, int Ncols) -> unsigned short* {
        unsigned short* p = (unsigned short*)alloc((size_t)Kp * Ncols * 2);
        const int total = Kp * Ncols;
        k_cvtW<<<dim3((total + 255) / 256), dim3(256), 0, stream>>>(F(leaf), p, K, Kp, Ncols);
        return p;
    };

    // --- param leaf indices (tree-flatten, sorted keys) ---
    // 0..6: atom_features, r, angle_h, src, dst, lsrc, ldst
    // 7: alignn[i] = {edge eggc, node eggc}, 14 leaves each, i-th at 7+i*28
    // eggc: +0 dgW +1 dgB +2 duW +3 duB +4 egW +5 egB +6 lne.b +7 lne.g
    //       +8 lnn.b +9 lnn.g +10 sgW +11 sgB +12 suW +13 suB
    // 119 ang_emb1{W,b,ln.b,ln.g} 123 ang_emb2 127 atom_emb 131 edge_emb1
    // 135 edge_emb2 139 fc.W 140 fc.b 141 gcn[i] at 141+14*i
    struct Eggc {
        const unsigned short *dgW, *duW, *egW, *sgW, *suW;
        const float *dgB, *duB, *egB, *sgB, *suB, *lneB, *lneG, *lnnB, *lnnG;
    };
    auto mkEggc = [&](int base) -> Eggc {
        Eggc E;
        E.dgW = cvtW(base + 0, 256, 256, 256);  E.dgB = F(base + 1);
        E.duW = cvtW(base + 2, 256, 256, 256);  E.duB = F(base + 3);
        E.egW = cvtW(base + 4, 256, 256, 256);  E.egB = F(base + 5);
        E.lneB = F(base + 6);  E.lneG = F(base + 7);
        E.lnnB = F(base + 8);  E.lnnG = F(base + 9);
        E.sgW = cvtW(base + 10, 256, 256, 256); E.sgB = F(base + 11);
        E.suW = cvtW(base + 12, 256, 256, 256); E.suB = F(base + 13);
        return E;
    };

    Eggc alE[4], alN[4], gcn[4];
    for (int i = 0; i < 4; i++) {
        alE[i] = mkEggc(7 + i * 28);
        alN[i] = mkEggc(7 + i * 28 + 14);
    }
    for (int i = 0; i < 4; i++) gcn[i] = mkEggc(141 + i * 14);

    unsigned short* angW1 = cvtW(119, 40, 64, 64);
    unsigned short* angW2 = cvtW(123, 64, 64, 256);
    unsigned short* atomW = cvtW(127, 92, 96, 256);
    unsigned short* edgW1 = cvtW(131, 80, 96, 64);
    unsigned short* edgW2 = cvtW(135, 64, 64, 256);

    auto gemm256 = [&](const unsigned short* A, const unsigned short* W, const float* bias,
                       const float* g, const float* be, float* oF, unsigned short* oB,
                       int M, int Kp, int mode) {
        dim3 grid((unsigned)(M / 32));
        if (mode == 1)
            k_gemm<256, 1><<<grid, 256, 0, stream>>>(A, W, bias, g, be, oF, oB, M, Kp);
        else
            k_gemm<256, 0><<<grid, 256, 0, stream>>>(A, W, bias, g, be, oF, oB, M, Kp);
    };
    auto gemm64 = [&](const unsigned short* A, const unsigned short* W, const float* bias,
                      const float* g, const float* be, unsigned short* oB, int M, int Kp) {
        k_gemm<64, 1><<<dim3((unsigned)(M / 32)), 256, 0, stream>>>(
            A, W, bias, g, be, (float*)nullptr, oB, M, Kp);
    };

    // ---- embeddings ----
    {
        const int total = NN * 96;
        k_cvtA<<<dim3((total + 255) / 256), 256, 0, stream>>>(F(0), atomb, NN, 92, 96);
    }
    gemm256(atomb, atomW, F(128), F(130), F(129), xf, xb, NN, 96, 1);      // atom_emb
    k_rbf_edge<<<dim3((NE + 255) / 256), 256, 0, stream>>>(F(1), rbfe, NE);
    gemm64 (rbfe, edgW1, F(132), F(134), F(133), e1b, NE, 96);             // edge_emb1
    gemm256(e1b,  edgW2, F(136), F(138), F(137), yf, yb, NE, 64, 1);       // edge_emb2
    k_rbf_angle<<<dim3((NT + 255) / 256), 256, 0, stream>>>(F(2), rbfa, NT);
    gemm64 (rbfa, angW1, F(120), F(122), F(121), a1b, NT, 64);             // ang_emb1
    gemm256(a1b,  angW2, F(124), F(126), F(125), zf, zb, NT, 64, 1);       // ang_emb2

    const int *src = I(3), *dst = I(4), *lsrc = I(5), *ldst = I(6);

    auto runEggc = [&](const Eggc& E, float* Xf, unsigned short* Xb, float* Yf,
                       unsigned short* Yb, const int* sI, const int* dI,
                       int rowsX, int rowsY) {
        gemm256(Xb, E.sgW, E.sgB, nullptr, nullptr, g0, nullptr, rowsX, 256, 0);
        gemm256(Xb, E.dgW, E.dgB, nullptr, nullptr, g1, nullptr, rowsX, 256, 0);
        gemm256(Xb, E.duW, E.duB, nullptr, nullptr, g2, nullptr, rowsX, 256, 0);
        gemm256(Xb, E.suW, E.suB, nullptr, nullptr, g3, nullptr, rowsX, 256, 0);
        gemm256(Yb, E.egW, E.egB, nullptr, nullptr, ce, nullptr, rowsY, 256, 0);
        const size_t zn = (size_t)rowsX * HID;
        k_zero<<<dim3((unsigned)((zn + 255) / 256)), 256, 0, stream>>>(sh, zn);
        k_zero<<<dim3((unsigned)((zn + 255) / 256)), 256, 0, stream>>>(ss, zn);
        k_edge<<<dim3((rowsY + 7) / 8), 256, 0, stream>>>(
            g0, g1, g2, ce, sI, dI, E.lneG, E.lneB, Yf, Yb, sh, ss, rowsY);
        k_node<<<dim3((rowsX + 7) / 8), 256, 0, stream>>>(
            g3, sh, ss, E.lnnG, E.lnnB, Xf, Xb, rowsX);
    };

    for (int i = 0; i < 4; i++) {
        runEggc(alN[i], xf, xb, yf, yb, src, dst, NN, NE);      // crystal graph
        runEggc(alE[i], yf, yb, zf, zb, lsrc, ldst, NE, NT);    // line graph
    }
    for (int i = 0; i < 4; i++) runEggc(gcn[i], xf, xb, yf, yb, src, dst, NN, NE);

    k_meanfc<<<dim3(1), 256, 0, stream>>>(xf, F(139), F(140), (float*)d_out, NN);
}
